// CMambaFiveHorizons_57612691309547
// MI455X (gfx1250) — matrix operand order repaired and checked by reading, not compiled, not run on hardware
//
#include <hip/hip_runtime.h>
#include <cmath>

typedef __attribute__((ext_vector_type(16))) _Float16 v16h;
typedef __attribute__((ext_vector_type(8)))  float    v8f;

namespace {
constexpr int cB = 4, cL = 1024, cF = 32, cDM = 256, cNL = 4, cDS = 16, cDC = 4, cNH = 4;
constexpr int cDI = 512, cDTR = 16;
constexpr int cM = cB * cL;           // 4096 rows of activations
}

// Branchless: load 8 consecutive f32 from base+(ok?off:0) -> 8 f16 slots, zeroed if !ok.
// For compile-time-true `ok` this folds to two b128 loads + cvt with no selects.
__device__ __forceinline__ void ld8cvt(_Float16* dst, const float* base, int off, bool ok) {
  const float* p = base + (ok ? off : 0);   // always in-bounds (row base valid for >=8)
  float4 f0 = *(const float4*)(p);
  float4 f1 = *(const float4*)(p + 4);
  const _Float16 z = (_Float16)0.0f;
  dst[0] = ok ? (_Float16)f0.x : z; dst[1] = ok ? (_Float16)f0.y : z;
  dst[2] = ok ? (_Float16)f0.z : z; dst[3] = ok ? (_Float16)f0.w : z;
  dst[4] = ok ? (_Float16)f1.x : z; dst[5] = ok ? (_Float16)f1.y : z;
  dst[6] = ok ? (_Float16)f1.z : z; dst[7] = ok ? (_Float16)f1.w : z;
}

// C[M,N] = act( A[M,K] (row-major, stride lda) @ W[N,K]^T + bias )
// ACT: 0 = none, 1 = +bias, 2 = +bias then softplus
// Each wave32 computes a 16 x (16*NB) strip: the A fragment is loaded once per
// K-step and reused across NB back-to-back WMMAs (halves A-side L2 traffic at NB=2).
template <int K, int ACT, int NB>
__global__ void wmma_gemm_f16(const float* __restrict__ A, int lda,
                              const float* __restrict__ W,
                              const float* __restrict__ bias,
                              float* __restrict__ C, int ldc,
                              int M, int N) {
  const int lane = threadIdx.x & 31;
  const int wave = threadIdx.x >> 5;
  const int tilesN = N / (16 * NB);
  const int tile = blockIdx.x * (blockDim.x >> 5) + wave;
  if (tile >= (M >> 4) * tilesN) return;          // wave-uniform exit (EXEC stays full)
  const int tm = (tile / tilesN) << 4;
  const int tn = (tile % tilesN) * (16 * NB);
  const int half = lane >> 4;
  const int lr = lane & 15;

  const float* arow = A + (size_t)(tm + lr) * lda;     // A row m = tm + lr (both halves)
  const float* wrow[NB];
#pragma unroll
  for (int j = 0; j < NB; ++j)
    wrow[j] = W + (size_t)(tn + 16 * j + lr) * K;      // W row n for each N-tile

  v8f acc[NB];
#pragma unroll
  for (int j = 0; j < NB; ++j) acc[j] = {};

  v16h a, b[NB];
  _Float16* ap = (_Float16*)&a;

#pragma unroll
  for (int k0 = 0; k0 < K; k0 += 32) {
    // A 16-bit layout: lane-half holds K = {k0+8h .. +7} and {k0+16+8h .. +7}
    const int ga0 = k0 + 8 * half;
    const int ga1 = k0 + 16 + 8 * half;
    // B 16-bit layout: same K order as A (ga0 .. +7, then ga1 .. +7)
    ld8cvt(ap,     arow, ga0, ga0 + 8 <= K);
    ld8cvt(ap + 8, arow, ga1, ga1 + 8 <= K);
#pragma unroll
    for (int j = 0; j < NB; ++j) {
      _Float16* bp = (_Float16*)&b[j];
      ld8cvt(bp,     wrow[j], ga0, ga0 + 8 <= K);
      ld8cvt(bp + 8, wrow[j], ga1, ga1 + 8 <= K);
    }
#pragma unroll
    for (int j = 0; j < NB; ++j)
      acc[j] = __builtin_amdgcn_wmma_f32_16x16x32_f16(
          false, a, false, b[j], (short)0, acc[j], false, false);
  }

  // D layout: n = lane%16, m = r + 8*half
#pragma unroll
  for (int j = 0; j < NB; ++j) {
    const int n = tn + 16 * j + lr;
    const float bv = (ACT != 0) ? bias[n] : 0.0f;
    const float* cp = (const float*)&acc[j];
#pragma unroll
    for (int r = 0; r < 8; ++r) {
      const int m = tm + r + 8 * half;
      float v = cp[r] + bv;
      if (ACT == 2) v = (v > 20.0f) ? v : log1pf(__expf(v));  // softplus
      C[(size_t)m * ldc + n] = v;
    }
  }
}

// Depthwise causal conv (DC=4) over L + bias + SiLU.  u = xz[..., :DI]
__global__ void conv_silu_kernel(const float* __restrict__ xz,
                                 const float* __restrict__ cw,   // (DI, DC)
                                 const float* __restrict__ cb,   // (DI,)
                                 float* __restrict__ uc) {
  const int idx = blockIdx.x * blockDim.x + threadIdx.x;
  if (idx >= cB * cL * cDI) return;
  const int d = idx % cDI;
  const int t = (idx / cDI) % cL;
  const int b = idx / (cDI * cL);
  float acc = cb[d];
#pragma unroll
  for (int k = 0; k < cDC; ++k) {
    const int tt = t + k - (cDC - 1);
    if (tt >= 0)
      acc += xz[((size_t)(b * cL + tt)) * (2 * cDI) + d] * cw[d * cDC + k];
  }
  uc[idx] = acc / (1.0f + __expf(-acc));   // SiLU
}

// Selective scan: one lane per (b,d) channel, 16-wide state in registers.
// Fuses y = scan + uc*Dp, gate by silu(z); result overwrites deltaBuf in place.
__global__ void scan_kernel(float* __restrict__ deltaBuf,        // (B,L,DI) in: delta, out: gated y
                            const float* __restrict__ uc,        // (B,L,DI)
                            const float* __restrict__ dbc,       // (B,L,48): [16:32)=Bc, [32:48)=Cc
                            const float* __restrict__ xz,        // (B,L,1024): z = [512:)
                            const float* __restrict__ A_log,     // (DI,DS)
                            const float* __restrict__ Dp) {      // (DI,)
  const int idx = blockIdx.x * blockDim.x + threadIdx.x;
  if (idx >= cB * cDI) return;
  const int d = idx % cDI;
  const int b = idx / cDI;

  float a[cDS], h[cDS];
#pragma unroll
  for (int n = 0; n < cDS; ++n) { a[n] = -__expf(A_log[d * cDS + n]); h[n] = 0.0f; }
  const float dp = Dp[d];

  for (int t = 0; t < cL; ++t) {
    const size_t row = (size_t)(b * cL + t);
    const float delta = deltaBuf[row * cDI + d];
    const float u     = uc[row * cDI + d];
    const float4* bc4 = (const float4*)(dbc + row * 48 + cDTR);
    const float4* cc4 = (const float4*)(dbc + row * 48 + cDTR + cDS);
    float bcv[cDS], ccv[cDS];
#pragma unroll
    for (int q = 0; q < cDS / 4; ++q) {
      float4 bb = bc4[q], cc = cc4[q];
      bcv[4 * q + 0] = bb.x; bcv[4 * q + 1] = bb.y; bcv[4 * q + 2] = bb.z; bcv[4 * q + 3] = bb.w;
      ccv[4 * q + 0] = cc.x; ccv[4 * q + 1] = cc.y; ccv[4 * q + 2] = cc.z; ccv[4 * q + 3] = cc.w;
    }
    const float du = delta * u;
    float y = 0.0f;
#pragma unroll
    for (int n = 0; n < cDS; ++n) {
      h[n] = __expf(delta * a[n]) * h[n] + du * bcv[n];
      y += h[n] * ccv[n];
    }
    const float z = xz[row * (2 * cDI) + cDI + d];
    const float g = z / (1.0f + __expf(-z));
    deltaBuf[row * cDI + d] = (y + u * dp) * g;
  }
}

// h[row] += LayerNorm(blk[row]) * g + beta ; one 256-thread block per row.
__global__ void ln_residual_kernel(const float* __restrict__ blk,
                                   const float* __restrict__ g,
                                   const float* __restrict__ beta,
                                   float* __restrict__ h) {
  __shared__ float s[cDM];
  const int row = blockIdx.x;
  const int t = threadIdx.x;
  const float x = blk[(size_t)row * cDM + t];
  s[t] = x;
  __syncthreads();
  for (int off = cDM / 2; off > 0; off >>= 1) {
    if (t < off) s[t] += s[t + off];
    __syncthreads();
  }
  const float mu = s[0] / (float)cDM;
  __syncthreads();
  const float dx = x - mu;
  s[t] = dx * dx;
  __syncthreads();
  for (int off = cDM / 2; off > 0; off >>= 1) {
    if (t < off) s[t] += s[t + off];
    __syncthreads();
  }
  const float var = s[0] / (float)cDM;
  h[(size_t)row * cDM + t] += dx * rsqrtf(var + 1e-5f) * g[t] + beta[t];
}

// p[b,dm] = mean over L of h[b,t,dm]
__global__ void mean_kernel(const float* __restrict__ h, float* __restrict__ p) {
  const int idx = blockIdx.x * blockDim.x + threadIdx.x;
  if (idx >= cB * cDM) return;
  const int dm = idx % cDM;
  const int b = idx / cDM;
  float s = 0.0f;
  for (int t = 0; t < cL; ++t) s += h[((size_t)(b * cL + t)) * cDM + dm];
  p[idx] = s / (float)cL;
}

__global__ void head1_kernel(const float* __restrict__ p,
                             const float* __restrict__ w1,
                             const float* __restrict__ b1,
                             float* __restrict__ hid) {
  const int idx = blockIdx.x * blockDim.x + threadIdx.x;
  if (idx >= cB * cDM) return;
  const int j = idx % cDM;
  const int b = idx / cDM;
  float s = b1[j];
  for (int k = 0; k < cDM; ++k) s += p[b * cDM + k] * w1[j * cDM + k];
  hid[idx] = fmaxf(s, 0.0f);
}

__global__ void head2_kernel(const float* __restrict__ hid,
                             const float* __restrict__ w2,
                             const float* __restrict__ b2,
                             float* __restrict__ out) {
  const int idx = blockIdx.x * blockDim.x + threadIdx.x;
  if (idx >= cB * cNH) return;
  const int n = idx % cNH;
  const int b = idx / cNH;
  float s = b2[n];
  for (int k = 0; k < cDM; ++k) s += hid[b * cDM + k] * w2[n * cDM + k];
  out[idx] = s;
}

template <int K, int ACT, int NB>
static inline void launch_gemm(const float* A, int lda, const float* W, const float* bias,
                               float* C, int ldc, int M, int N, hipStream_t stream) {
  const int tiles = (M / 16) * (N / (16 * NB));
  dim3 grid((tiles + 7) / 8), blk(256);
  wmma_gemm_f16<K, ACT, NB><<<grid, blk, 0, stream>>>(A, lda, W, bias, C, ldc, M, N);
}

extern "C" void kernel_launch(void* const* d_in, const int* in_sizes, int n_in,
                              void* d_out, int out_size, void* d_ws, size_t ws_size,
                              hipStream_t stream) {
  (void)in_sizes; (void)n_in; (void)out_size; (void)ws_size;
  const float* x        = (const float*)d_in[0];
  const float* proj_w   = (const float*)d_in[1];
  const float* proj_b   = (const float*)d_in[2];
  const float* inproj_w = (const float*)d_in[3];
  const float* conv_w   = (const float*)d_in[4];
  const float* conv_b   = (const float*)d_in[5];
  const float* xproj_w  = (const float*)d_in[6];
  const float* dtproj_w = (const float*)d_in[7];
  const float* dtproj_b = (const float*)d_in[8];
  const float* A_log    = (const float*)d_in[9];
  const float* Dp       = (const float*)d_in[10];
  const float* outproj_w= (const float*)d_in[11];
  const float* ln_g     = (const float*)d_in[12];
  const float* ln_b     = (const float*)d_in[13];
  const float* head_w1  = (const float*)d_in[14];
  const float* head_b1  = (const float*)d_in[15];
  const float* head_w2  = (const float*)d_in[16];
  const float* head_b2  = (const float*)d_in[17];
  float* out = (float*)d_out;

  // Workspace carve-up (floats)
  float* ws = (float*)d_ws;
  size_t off = 0;
  float* hbuf   = ws + off; off += (size_t)cM * cDM;        // 4096*256
  float* xz     = ws + off; off += (size_t)cM * 2 * cDI;    // 4096*1024
  float* ucb    = ws + off; off += (size_t)cM * cDI;        // 4096*512
  float* dbc    = ws + off; off += (size_t)cM * (cDTR + 2 * cDS); // 4096*48
  float* delta  = ws + off; off += (size_t)cM * cDI;        // 4096*512 (reused as gate)
  float* blkout = ws + off; off += (size_t)cM * cDM;        // 4096*256
  float* pbuf   = ws + off; off += (size_t)cB * cDM;
  float* hidbuf = ws + off; off += (size_t)cB * cDM;

  // h = x @ proj_w.T + proj_b      (M=4096, K=32, N=256)
  launch_gemm<cF, 1, 2>(x, cF, proj_w, proj_b, hbuf, cDM, cM, cDM, stream);

  for (int i = 0; i < cNL; ++i) {
    const float* inw  = inproj_w + (size_t)i * (2 * cDI) * cDM;
    const float* cw   = conv_w   + (size_t)i * cDI * cDC;
    const float* cb   = conv_b   + (size_t)i * cDI;
    const float* xpw  = xproj_w  + (size_t)i * (cDTR + 2 * cDS) * cDI;
    const float* dtw  = dtproj_w + (size_t)i * cDI * cDTR;
    const float* dtb  = dtproj_b + (size_t)i * cDI;
    const float* Al   = A_log    + (size_t)i * cDI * cDS;
    const float* Dpl  = Dp       + (size_t)i * cDI;
    const float* ow   = outproj_w+ (size_t)i * cDM * cDI;
    const float* lg   = ln_g     + (size_t)i * cDM;
    const float* lb   = ln_b     + (size_t)i * cDM;

    // xz = h @ inw.T               (K=256, N=1024)
    launch_gemm<cDM, 0, 2>(hbuf, cDM, inw, nullptr, xz, 2 * cDI, cM, 2 * cDI, stream);
    // uc = silu(causal_conv(u) + cb)
    conv_silu_kernel<<<(cM * cDI + 255) / 256, 256, 0, stream>>>(xz, cw, cb, ucb);
    // dbc = uc @ xpw.T             (K=512, N=48; 48 % 32 != 0 -> NB=1)
    launch_gemm<cDI, 0, 1>(ucb, cDI, xpw, nullptr, dbc, cDTR + 2 * cDS, cM, cDTR + 2 * cDS, stream);
    // delta = softplus(dt @ dtw.T + dtb)   (A = dbc[:, :16], lda=48, K=16, N=512)
    launch_gemm<cDTR, 2, 2>(dbc, cDTR + 2 * cDS, dtw, dtb, delta, cDI, cM, cDI, stream);
    // selective scan + D skip + silu(z) gate (in-place into delta)
    scan_kernel<<<(cB * cDI + 255) / 256, 256, 0, stream>>>(delta, ucb, dbc, xz, Al, Dpl);
    // blkout = gate @ ow.T         (K=512, N=256)
    launch_gemm<cDI, 0, 2>(delta, cDI, ow, nullptr, blkout, cDM, cM, cDM, stream);
    // h += LayerNorm(blkout)
    ln_residual_kernel<<<cM, cDM, 0, stream>>>(blkout, lg, lb, hbuf);
  }

  mean_kernel<<<(cB * cDM + 255) / 256, 256, 0, stream>>>(hbuf, pbuf);
  head1_kernel<<<(cB * cDM + 255) / 256, 256, 0, stream>>>(pbuf, head_w1, head_b1, hidbuf);
  head2_kernel<<<1, 32, 0, stream>>>(hidbuf, head_w2, head_b2, out);
}
